// Hippocampus_41094247088331
// MI455X (gfx1250) — matrix operand order repaired and checked by reading, not compiled, not run on hardware
//
#include <hip/hip_runtime.h>
#include <hip/hip_bf16.h>
#include <math.h>

typedef _Float16 v16h __attribute__((ext_vector_type(16)));
typedef _Float16 v8h  __attribute__((ext_vector_type(8)));
typedef float    v8f  __attribute__((ext_vector_type(8)));

#define T_STEPS   1024
#define I_DIM     1024
#define H_DIM     2048
#define NWG       64
#define NTHREADS  256   // 8 wave32 waves per WG

__device__ __forceinline__ float sigm(float x) { return 1.0f / (1.0f + expf(-x)); }

__device__ __forceinline__ v8f zero8() {
  v8f z;
#pragma unroll
  for (int i = 0; i < 8; ++i) z[i] = 0.0f;
  return z;
}

__device__ __forceinline__ v16h cat16(v8h lo, v8h hi) {
  return __builtin_shufflevector(lo, hi, 0, 1, 2, 3, 4, 5, 6, 7,
                                 8, 9, 10, 11, 12, 13, 14, 15);
}

// Weight B-fragment: pre-packed so lane L reads halves [L*8,L*8+8) and
// [256+L*8, ...) -> two contiguous global_load_b128 per lane, pure VMEM.
__device__ __forceinline__ v16h load_wfrag(const _Float16* __restrict__ Wp,
                                           int ktiles, int rt, int kc, int lane) {
  const _Float16* tp = Wp + (((size_t)rt * (size_t)ktiles + (size_t)kc) << 9);
  v8h lo = *(const v8h*)(tp + (lane << 3));
  v8h hi = *(const v8h*)(tp + 256 + (lane << 3));
  return cat16(lo, hi);
}

// x A-fragment (16x32, only row M=0 live): lane 0 holds K {0..7,16..23},
// lane 16 holds K {8..15,24..31}; all other lanes load from a zeroed buffer
// so the fragment is built entirely by VMEM (no VALU writes -> no WMMA WAR nops).
__device__ __forceinline__ v16h load_xfrag(const _Float16* __restrict__ chunk,
                                           const _Float16* __restrict__ zbuf,
                                           int lane) {
  const _Float16* p = ((lane & 15) == 0) ? (chunk + ((lane >> 4) << 3)) : zbuf;
  v8h lo = *(const v8h*)(p);
  v8h hi = *(const v8h*)(p + 16);
  return cat16(lo, hi);
}

__device__ __forceinline__ void grid_sync(unsigned* cnt, unsigned& target) {
  __threadfence();
  __syncthreads();
  target += NWG;
  if (threadIdx.x == 0) {
    __hip_atomic_fetch_add(cnt, 1u, __ATOMIC_ACQ_REL, __HIP_MEMORY_SCOPE_AGENT);
    while (__hip_atomic_load(cnt, __ATOMIC_ACQUIRE, __HIP_MEMORY_SCOPE_AGENT) < target) {
      __builtin_amdgcn_s_sleep(1);
    }
  }
  __syncthreads();
  __threadfence();
}

// LSTM phase: 128 row tiles, 3 gate matrices (i,g,o) share one x A-fragment.
// 4-way K split (waves 0-3 -> tile wg*2, waves 4-7 -> wg*2+1), LDS reduce.
__device__ __forceinline__ void lstm_phase(const _Float16* __restrict__ Wi,
                                           const _Float16* __restrict__ Wg,
                                           const _Float16* __restrict__ Wo,
                                           const float* __restrict__ bih,
                                           const float* __restrict__ bhh,
                                           const _Float16* __restrict__ xin,  // [2048] fp16
                                           const _Float16* __restrict__ zbuf,
                                           _Float16* __restrict__ hout,       // [H] fp16
                                           float (*red)[48],
                                           int wg, int wave, int lane) {
  const int kt = 64;                          // K=2048 -> 64 chunks of 32
  const int rt = wg * 2 + (wave >> 2);        // 128 row tiles
  const int sub = wave & 3;
  v8f ai = zero8(), ag = zero8(), ao = zero8();
  const int c0 = sub * 16;
#pragma unroll 4
  for (int u = 0; u < 16; ++u) {
    const int c = c0 + u;
    v16h a = load_xfrag(xin + c * 32, zbuf, lane);
    v16h b0 = load_wfrag(Wi, kt, rt, c, lane);
    ai = __builtin_amdgcn_wmma_f32_16x16x32_f16(false, a, false, b0, (short)0, ai, false, false);
    v16h b1 = load_wfrag(Wg, kt, rt, c, lane);
    ag = __builtin_amdgcn_wmma_f32_16x16x32_f16(false, a, false, b1, (short)0, ag, false, false);
    v16h b2 = load_wfrag(Wo, kt, rt, c, lane);
    ao = __builtin_amdgcn_wmma_f32_16x16x32_f16(false, a, false, b2, (short)0, ao, false, false);
  }
  if (lane < 16) {                            // D row M=0: N=lane in acc[0]
    red[wave][lane]      = ai[0];
    red[wave][16 + lane] = ag[0];
    red[wave][32 + lane] = ao[0];
  }
  __syncthreads();
  if (sub == 0 && lane < 16) {
    const int row = rt * 16 + lane;
    float si = 0.f, sg = 0.f, so = 0.f;
    const int g0 = wave;                      // 0 or 4
#pragma unroll
    for (int w = 0; w < 4; ++w) {
      si += red[g0 + w][lane];
      sg += red[g0 + w][16 + lane];
      so += red[g0 + w][32 + lane];
    }
    si += bih[row]             + bhh[row];               // i gate rows [0,H)
    sg += bih[2 * H_DIM + row] + bhh[2 * H_DIM + row];   // g gate rows [2H,3H)
    so += bih[3 * H_DIM + row] + bhh[3 * H_DIM + row];   // o gate rows [3H,4H)
    float c = sigm(si) * tanhf(sg);                       // c0 = 0 -> f gate dead
    hout[row] = (_Float16)(sigm(so) * tanhf(c));
  }
}

__global__ __launch_bounds__(NTHREADS, 1)
void rnn_scan(const _Float16* __restrict__ W1p,
              const _Float16* __restrict__ Wi1, const _Float16* __restrict__ Wg1,
              const _Float16* __restrict__ Wo1,
              const _Float16* __restrict__ Wi2, const _Float16* __restrict__ Wg2,
              const _Float16* __restrict__ Wo2,
              const _Float16* __restrict__ W2p,
              const float* __restrict__ b1,
              const float* __restrict__ bih1, const float* __restrict__ bhh1,
              const float* __restrict__ bih2, const float* __restrict__ bhh2,
              const float* __restrict__ b2,
              const _Float16* __restrict__ Xh,    // [T,I] fp16
              const _Float16* __restrict__ zbuf,  // 32 zero halves
              _Float16* __restrict__ vin2,        // [2I]: [x1 ; out_prev]
              _Float16* __restrict__ h1h,         // [H]
              _Float16* __restrict__ h2h,         // [H]
              float* __restrict__ out,            // [T,I] fp32
              unsigned* __restrict__ cnt) {
  const int wg   = blockIdx.x;
  const int wave = (int)(threadIdx.x >> 5);
  const int lane = (int)(threadIdx.x & 31);
  __shared__ float red[8][48];
  unsigned target = 0;

  for (int t = 0; t < T_STEPS; ++t) {
    // -------- Phase 1: x1 = W1 . [h2_prev ; x_t] + b1  (1024 rows, K=3072) --------
    {
      const int rt = wg;                       // 64 tiles
      const int kt = (H_DIM + I_DIM) / 32;     // 96 chunks, 8-way K split: 12/wave
      v8f acc = zero8();
      const int c0 = wave * 12;
#pragma unroll 4
      for (int u = 0; u < 12; ++u) {
        const int c = c0 + u;
        const _Float16* src = (c < H_DIM / 32) ? (h2h + c * 32)
                                               : (Xh + (size_t)t * I_DIM + (c - H_DIM / 32) * 32);
        v16h a = load_xfrag(src, zbuf, lane);
        v16h b = load_wfrag(W1p, kt, rt, c, lane);
        acc = __builtin_amdgcn_wmma_f32_16x16x32_f16(false, a, false, b, (short)0, acc, false, false);
      }
      if (lane < 16) red[wave][lane] = acc[0];
      __syncthreads();
      if (wave == 0 && lane < 16) {
        float s = b1[rt * 16 + lane];
#pragma unroll
        for (int w = 0; w < 8; ++w) s += red[w][lane];
        vin2[rt * 16 + lane] = (_Float16)s;    // x1 (linear, no activation)
      }
    }
    grid_sync(cnt, target);

    // -------- Phase 2: h1 = lstm_cell(W_ih1, [x1 ; out_prev]) --------
    lstm_phase(Wi1, Wg1, Wo1, bih1, bhh1, vin2, zbuf, h1h, red, wg, wave, lane);
    grid_sync(cnt, target);

    // -------- Phase 3: h2 = lstm_cell(W_ih2, h1) --------
    lstm_phase(Wi2, Wg2, Wo2, bih2, bhh2, h1h, zbuf, h2h, red, wg, wave, lane);
    grid_sync(cnt, target);

    // -------- Phase 4: o = sigmoid(W2 . h2 + b2)  (1024 rows, K=2048) --------
    {
      const int rt = wg;
      const int kt = H_DIM / 32;               // 64 chunks, 8-way split: 8/wave
      v8f acc = zero8();
      const int c0 = wave * 8;
#pragma unroll 4
      for (int u = 0; u < 8; ++u) {
        const int c = c0 + u;
        v16h a = load_xfrag(h2h + c * 32, zbuf, lane);
        v16h b = load_wfrag(W2p, kt, rt, c, lane);
        acc = __builtin_amdgcn_wmma_f32_16x16x32_f16(false, a, false, b, (short)0, acc, false, false);
      }
      if (lane < 16) red[wave][lane] = acc[0];
      __syncthreads();
      if (wave == 0 && lane < 16) {
        const int row = rt * 16 + lane;
        float s = b2[row];
#pragma unroll
        for (int w = 0; w < 8; ++w) s += red[w][lane];
        const float o = sigm(s);
        out[(size_t)t * I_DIM + row] = o;      // fp32 result
        vin2[I_DIM + row] = (_Float16)o;       // out_prev for next step
      }
    }
    grid_sync(cnt, target);
  }
}

// Repack row-major fp32 weights into fp16 WMMA B-fragment tiles (512 halves/tile).
// Lane L reads halves [L*8,L*8+8) (k=8*(L/16)+s) and [256+L*8,...) (k=16+8*(L/16)+s),
// packed value = W[rt*16 + (L%16)][kc*32 + k]  (transposed 16x32 -> 32x16 tile).
__global__ void pack_w(const float* __restrict__ src, _Float16* __restrict__ dst,
                       int R, int K, int rowOff) {
  const int ktiles = K >> 5;
  const size_t total = (size_t)R * (size_t)K;
  for (size_t e = (size_t)blockIdx.x * blockDim.x + threadIdx.x; e < total;
       e += (size_t)gridDim.x * blockDim.x) {
    const size_t ti = e >> 9;
    const int p = (int)(e & 511);
    const int rt = (int)(ti / ktiles);
    const int kc = (int)(ti % ktiles);
    int L, n;
    if (p < 256) { const int Ln = p >> 3; n = Ln & 15; L = 8 * (Ln >> 4) + (p & 7); }
    else         { const int q = p - 256; const int Ln = q >> 3; n = Ln & 15; L = 16 + 8 * (Ln >> 4) + (q & 7); }
    const int row = rt * 16 + n;
    const int col = kc * 32 + L;
    dst[e] = (_Float16)src[(size_t)(row + rowOff) * (size_t)K + col];
  }
}

__global__ void cvt_f16(const float* __restrict__ s, _Float16* __restrict__ d, int n) {
  for (int i = blockIdx.x * blockDim.x + threadIdx.x; i < n; i += gridDim.x * blockDim.x)
    d[i] = (_Float16)s[i];
}

__global__ void init_state(unsigned* cnt, _Float16* vin2, _Float16* h1, _Float16* h2,
                           _Float16* zbuf) {
  if (threadIdx.x == 0) *cnt = 0u;
  for (int i = threadIdx.x; i < 2 * I_DIM; i += blockDim.x) vin2[i] = (_Float16)0.0f;
  for (int i = threadIdx.x; i < H_DIM; i += blockDim.x) { h1[i] = (_Float16)0.0f; h2[i] = (_Float16)0.0f; }
  if (threadIdx.x < 32) zbuf[threadIdx.x] = (_Float16)0.0f;
}

extern "C" void kernel_launch(void* const* d_in, const int* in_sizes, int n_in,
                              void* d_out, int out_size, void* d_ws, size_t ws_size,
                              hipStream_t stream) {
  (void)in_sizes; (void)n_in; (void)out_size; (void)ws_size;
  const float* X    = (const float*)d_in[0];
  const float* W1   = (const float*)d_in[1];
  const float* b1   = (const float*)d_in[2];
  const float* Wih1 = (const float*)d_in[3];
  const float* bih1 = (const float*)d_in[4];
  const float* bhh1 = (const float*)d_in[5];
  const float* Wih2 = (const float*)d_in[6];
  const float* bih2 = (const float*)d_in[7];
  const float* bhh2 = (const float*)d_in[8];
  const float* W2   = (const float*)d_in[9];
  const float* b2   = (const float*)d_in[10];
  float* out = (float*)d_out;

  char* ws = (char*)d_ws;
  size_t off = 0;
  auto take = [&](size_t bytes) -> char* {
    char* p = ws + off;
    off = (off + bytes + 255) & ~(size_t)255;
    return p;
  };
  unsigned* cnt  = (unsigned*)take(sizeof(unsigned));
  _Float16* zbuf = (_Float16*)take((size_t)32 * 2);
  _Float16* vin2 = (_Float16*)take((size_t)2 * I_DIM * 2);
  _Float16* h1h  = (_Float16*)take((size_t)H_DIM * 2);
  _Float16* h2h  = (_Float16*)take((size_t)H_DIM * 2);
  _Float16* Xh   = (_Float16*)take((size_t)T_STEPS * I_DIM * 2);
  _Float16* W1p  = (_Float16*)take((size_t)I_DIM * (H_DIM + I_DIM) * 2);
  _Float16* Wi1  = (_Float16*)take((size_t)H_DIM * (2 * I_DIM) * 2);
  _Float16* Wg1  = (_Float16*)take((size_t)H_DIM * (2 * I_DIM) * 2);
  _Float16* Wo1  = (_Float16*)take((size_t)H_DIM * (2 * I_DIM) * 2);
  _Float16* Wi2  = (_Float16*)take((size_t)H_DIM * H_DIM * 2);
  _Float16* Wg2  = (_Float16*)take((size_t)H_DIM * H_DIM * 2);
  _Float16* Wo2  = (_Float16*)take((size_t)H_DIM * H_DIM * 2);
  _Float16* W2p  = (_Float16*)take((size_t)I_DIM * H_DIM * 2);

  init_state<<<1, 256, 0, stream>>>(cnt, vin2, h1h, h2h, zbuf);
  cvt_f16<<<1024, 256, 0, stream>>>(X, Xh, T_STEPS * I_DIM);

  pack_w<<<2048, 256, 0, stream>>>(W1,   W1p, I_DIM, H_DIM + I_DIM, 0);
  pack_w<<<2048, 256, 0, stream>>>(Wih1, Wi1, H_DIM, 2 * I_DIM, 0);
  pack_w<<<2048, 256, 0, stream>>>(Wih1, Wg1, H_DIM, 2 * I_DIM, 2 * H_DIM);
  pack_w<<<2048, 256, 0, stream>>>(Wih1, Wo1, H_DIM, 2 * I_DIM, 3 * H_DIM);
  pack_w<<<2048, 256, 0, stream>>>(Wih2, Wi2, H_DIM, H_DIM, 0);
  pack_w<<<2048, 256, 0, stream>>>(Wih2, Wg2, H_DIM, H_DIM, 2 * H_DIM);
  pack_w<<<2048, 256, 0, stream>>>(Wih2, Wo2, H_DIM, H_DIM, 3 * H_DIM);
  pack_w<<<2048, 256, 0, stream>>>(W2,   W2p, I_DIM, H_DIM, 0);

  rnn_scan<<<NWG, NTHREADS, 0, stream>>>(W1p, Wi1, Wg1, Wo1, Wi2, Wg2, Wo2, W2p,
                                         b1, bih1, bhh1, bih2, bhh2, b2,
                                         Xh, zbuf, vin2, h1h, h2h, out, cnt);
}
